// NAIS_regionEmbedding_65240553226372
// MI455X (gfx1250) — matrix operand order repaired and checked by reading, not compiled, not run on hardware
//
#include <hip/hip_runtime.h>
#include <math.h>

// Problem constants (match reference)
#define B_   1024
#define H_   256
#define EMB  128
#define D2   256   // 2*EMBED (K of the GEMM)
#define HID  256   // hidden (N of the GEMM)
#define CH   64    // history rows per chunk
#define NCH  4     // chunks (CH*NCH == H_)
#define SI   264   // LDS stride (bf16 elems) for inp tile  (256 + pad)
#define SW   264   // LDS stride (bf16 elems) for W1^T

typedef __attribute__((ext_vector_type(16))) __bf16 v16bf;
typedef __attribute__((ext_vector_type(8)))  float  v8f;

union Frag16 { v16bf v; unsigned int u[8]; };

__device__ __forceinline__ unsigned int f2bf(float f) {
    unsigned int u = __builtin_bit_cast(unsigned int, f);
    u += 0x7fffu + ((u >> 16) & 1u);            // round-to-nearest-even
    return u >> 16;
}

__device__ __forceinline__ void load_frags(Frag16& fa, Frag16& fb,
                                           const unsigned short* arow,
                                           const unsigned short* brow, int kk) {
    // A (16x32 bf16): lane M=l%16, Kbase = kk*32 + (l/16)*8 ; v4-7 at +16
    *(uint4*)&fa.u[0] = *(const uint4*)(arow + kk * 32);
    *(uint4*)&fa.u[4] = *(const uint4*)(arow + kk * 32 + 16);
    // B (32x16 bf16): lane N=l%16, Kbase = kk*32 + (l/16)*8 ; v4-7 at +16
    *(uint4*)&fb.u[0] = *(const uint4*)(brow + kk * 32);
    *(uint4*)&fb.u[4] = *(const uint4*)(brow + kk * 32 + 16);
}

// ---------------------------------------------------------------------------
// Stage 1: tgt[b, 0:256] = concat(E_targ[target[b]], E_reg[target_region[b]])
// ---------------------------------------------------------------------------
__global__ void build_tgt_kernel(const int* __restrict__ target,
                                 const int* __restrict__ target_region,
                                 const float* __restrict__ E_targ,
                                 const float* __restrict__ E_reg,
                                 float* __restrict__ tgt) {
    const int b = blockIdx.x;
    const int d = threadIdx.x;
    float v = (d < EMB) ? E_targ[(long)target[b] * EMB + d]
                        : E_reg [(long)target_region[b] * EMB + (d - EMB)];
    tgt[b * D2 + d] = v;
}

// ---------------------------------------------------------------------------
// Stage 2: one workgroup per batch element (256 threads = 8 wave32)
// ---------------------------------------------------------------------------
__global__ void nais_main_kernel(const int*   __restrict__ history,
                                 const int*   __restrict__ target,
                                 const int*   __restrict__ history_region,
                                 const float* __restrict__ E_hist,
                                 const float* __restrict__ E_reg,
                                 const float* __restrict__ W1,
                                 const float* __restrict__ b1,
                                 const float* __restrict__ w2,
                                 const float* __restrict__ tgt,
                                 float*       __restrict__ out) {
    extern __shared__ char smem[];
    unsigned short* sW   = (unsigned short*)smem;     // [HID][SW]  W1^T bf16 (N-major, K inner)
    unsigned short* sIn  = sW + HID * SW;             // [CH][SI]   inp tile bf16
    float* sTgt   = (float*)(sIn + CH * SI);          // [D2]
    float* sB1    = sTgt + D2;                        // [HID]
    float* sW2    = sB1 + HID;                        // [HID]
    float* sScore = sW2 + HID;                        // [H_]
    float* sDot   = sScore + H_;                      // [H_]
    float* sRed   = sDot + H_;                        // [16]

    const int b    = blockIdx.x;
    const int t    = threadIdx.x;
    const int lane = t & 31;
    const int wv   = t >> 5;

    // Stage constants into LDS (one element per thread each).
    sTgt[t]   = tgt[b * D2 + t];
    sB1[t]    = b1[t];
    sW2[t]    = w2[t];
    sScore[t] = 0.0f;

    // W1 [K=D2][N=HID] fp32 -> LDS transposed bf16 sW[n][k].
    // Thread t owns column n = t: global reads fully coalesced across lanes,
    // LDS writes are packed ds_store_b128 (8 bf16 per store).
    {
        unsigned short* dst = sW + t * SW;
        for (int k0 = 0; k0 < D2; k0 += 8) {
            float w[8];
            #pragma unroll
            for (int kk = 0; kk < 8; ++kk)
                w[kk] = W1[(long)(k0 + kk) * HID + t];
            uint4 pk;
            pk.x = f2bf(w[0]) | (f2bf(w[1]) << 16);
            pk.y = f2bf(w[2]) | (f2bf(w[3]) << 16);
            pk.z = f2bf(w[4]) | (f2bf(w[5]) << 16);
            pk.w = f2bf(w[6]) | (f2bf(w[7]) << 16);
            *(uint4*)(dst + k0) = pk;
        }
    }
    __syncthreads();

    const int tval = target[b];
    const int half = lane >> 4;     // 0: lanes 0-15, 1: lanes 16-31
    const int l16  = lane & 15;
    const int r    = t >> 2;        // gather: row owned by this thread
    const int q    = t & 3;         //         quarter (64 cols) of that row

    for (int c = 0; c < NCH; ++c) {
        // ---- Gather + broadcast-multiply + in-register rowdot ------------
        {
            const int h   = c * CH + r;
            const int idx = (q < 2) ? history[b * H_ + h]
                                    : history_region[b * H_ + h];
            const float4* src4 = (q < 2)
                ? (const float4*)(E_hist + (long)idx * EMB + q * 64)
                : (const float4*)(E_reg  + (long)idx * EMB + (q - 2) * 64);
            const float4* tgt4 = (const float4*)(sTgt + q * 64);
            unsigned short* drow = sIn + r * SI + q * 64;
            float rd = 0.f;
            #pragma unroll
            for (int j = 0; j < 16; ++j) {
                float4 a = src4[j];
                float4 g = tgt4[j];
                float p0 = a.x * g.x, p1 = a.y * g.y;
                float p2 = a.z * g.z, p3 = a.w * g.w;
                rd += (p0 + p1) + (p2 + p3);
                uint2 pk;
                pk.x = f2bf(p0) | (f2bf(p1) << 16);
                pk.y = f2bf(p2) | (f2bf(p3) << 16);
                *(uint2*)(drow + j * 4) = pk;          // ds_store_b64
            }
            rd += __shfl_xor(rd, 1, 32);
            rd += __shfl_xor(rd, 2, 32);
            if (q == 0) sDot[h] = rd;
        }
        // Prefetch next chunk's history-embedding rows (global_prefetch_b8).
        if (c + 1 < NCH && t < CH) {
            int nit = history[b * H_ + (c + 1) * CH + t];
            __builtin_prefetch(&E_hist[(long)nit * EMB], 0, 1);
        }
        __syncthreads();

        // ---- GEMM: inp[CH x D2] x W1t -> relu(+b1) -> dot(w2) -> scores --
        for (int m = 0; m < CH / 16; ++m) {
            float ps[8] = {0.f, 0.f, 0.f, 0.f, 0.f, 0.f, 0.f, 0.f};
            #pragma unroll
            for (int nn = 0; nn < 2; ++nn) {
                const int nt = wv * 2 + nn;       // N-tile owned by this wave
                const int n  = nt * 16 + l16;     // this lane's output column
                const unsigned short* arow = sIn + (m * 16 + l16) * SI + half * 8;
                const unsigned short* brow = sW + n * SW + half * 8;

                Frag16 fa[2], fb[2];
                load_frags(fa[0], fb[0], arow, brow, 0);
                v8f acc = {};
                #pragma unroll
                for (int kk = 0; kk < D2 / 32; ++kk) {
                    const int cur = kk & 1;
                    if (kk + 1 < D2 / 32)          // software pipeline: load
                        load_frags(fa[cur ^ 1], fb[cur ^ 1], arow, brow, kk + 1);
                    acc = __builtin_amdgcn_wmma_f32_16x16x32_bf16(
                              false, fa[cur].v, false, fb[cur].v,
                              (short)0, acc, false, false);
                }
                // D lane holds (N = n, M = v + 8*half) of rows m*16+...
                const float bias = sB1[n];
                const float w2n  = sW2[n];
                #pragma unroll
                for (int v = 0; v < 8; ++v)
                    ps[v] += __builtin_fmaxf(acc[v] + bias, 0.f) * w2n;
            }
            // Reduce each row-partial over the 16 n-lanes of this half,
            // then one LDS float atomic per (row, half).
            #pragma unroll
            for (int v = 0; v < 8; ++v) {
                float p = ps[v];
                p += __shfl_xor(p, 1, 32);
                p += __shfl_xor(p, 2, 32);
                p += __shfl_xor(p, 4, 32);
                p += __shfl_xor(p, 8, 32);
                if (l16 == 0)
                    atomicAdd(&sScore[c * CH + m * 16 + v + half * 8], p);
            }
        }
        __syncthreads();
    }

    // ---- Final: masked exp, denom = sum^0.5, pred, sigmoid ---------------
    {
        float sc = sScore[t];
        float e  = (history[b * H_ + t] != tval) ? __expf(sc) : 0.0f;
        float p  = e * sDot[t];
        #pragma unroll
        for (int off = 16; off >= 1; off >>= 1) {
            e += __shfl_xor(e, off, 32);
            p += __shfl_xor(p, off, 32);
        }
        if (lane == 0) { sRed[wv * 2] = e; sRed[wv * 2 + 1] = p; }
    }
    __syncthreads();
    if (t == 0) {
        float S = 0.f, P = 0.f;
        for (int w = 0; w < 8; ++w) { S += sRed[w * 2]; P += sRed[w * 2 + 1]; }
        float pred = P / sqrtf(S);                  // BETA = 0.5
        out[b] = 1.0f / (1.0f + __expf(-pred));
    }
}

// ---------------------------------------------------------------------------
extern "C" void kernel_launch(void* const* d_in, const int* in_sizes, int n_in,
                              void* d_out, int out_size, void* d_ws, size_t ws_size,
                              hipStream_t stream) {
    const int*   history        = (const int*)  d_in[0];
    const int*   target         = (const int*)  d_in[1];
    const int*   history_region = (const int*)  d_in[2];
    const int*   target_region  = (const int*)  d_in[3];
    const float* E_hist         = (const float*)d_in[4];
    const float* E_targ         = (const float*)d_in[5];
    const float* E_reg          = (const float*)d_in[6];
    const float* W1             = (const float*)d_in[7];
    const float* b1             = (const float*)d_in[8];
    const float* w2             = (const float*)d_in[9];
    float* out    = (float*)d_out;
    float* tgtbuf = (float*)d_ws;                  // [B_, D2] fp32

    const size_t smem = (size_t)(HID * SW + CH * SI) * sizeof(unsigned short)
                      + (size_t)(D2 + HID + HID + H_ + H_) * sizeof(float)
                      + 16 * sizeof(float);        // ~174.2 KB < 320 KB/WGP

    hipFuncSetAttribute((const void*)nais_main_kernel,
                        hipFuncAttributeMaxDynamicSharedMemorySize, (int)smem);

    build_tgt_kernel<<<B_, D2, 0, stream>>>(target, target_region, E_targ, E_reg, tgtbuf);
    nais_main_kernel<<<B_, 256, smem, stream>>>(history, target, history_region,
                                                E_hist, E_reg, W1, b1, w2,
                                                tgtbuf, out);
}
